// MultiHeadAttention_644245094724
// MI455X (gfx1250) — matrix operand order repaired and checked by reading, not compiled, not run on hardware
//
#include <hip/hip_runtime.h>

#define DEVI __device__ __forceinline__

typedef _Float16 f16;
typedef __attribute__((ext_vector_type(16))) _Float16 v16h;
typedef __attribute__((ext_vector_type(8)))  float    v8f;

constexpr int BATCH = 4, SEQ = 2048, HID = 1024, NHEAD = 16, HDIM = 64;
constexpr int MTOT = BATCH * SEQ;   // 8192 rows of x / ctx

// ---------------- CDNA5 async copy helpers ----------------
// Low 32 bits of a generic pointer into LDS = LDS byte offset (ISA 10.2 aperture rules).
DEVI unsigned lds_off(const void* p) { return (unsigned)(unsigned long long)p; }

// GLOBAL_LOAD_ASYNC_TO_LDS_B128, GVS mode: mem = SGPR base + 32-bit VGPR offset.
// LDS dest address comes from the first VGPR operand. Tracked by ASYNCcnt.
DEVI void async_ld_lds_b128(unsigned lds_byte_off, unsigned glb_byte_off, const void* base) {
  asm volatile("global_load_async_to_lds_b128 %0, %1, %2 offset:0"
               :: "v"(lds_byte_off), "v"(glb_byte_off), "s"(base) : "memory");
}
DEVI void wait_async0() { asm volatile("s_wait_asynccnt 0" ::: "memory"); }

// ---------------- WMMA fragment helpers (wave32, 16x16x32 f16) ----------------
// A-matrix 16x32 (MxK): lane<16 -> row=lane, K in {0..15}; lane>=16 -> K in {16..31}
DEVI v16h ldsA_frag(const f16* p, int row0, int stride) {
  const int lane = threadIdx.x & 31;
  const int half = lane >> 4, lr = lane & 15;
  const f16* rp = p + (size_t)(row0 + lr) * stride + half * 16;
  v16h a;
#pragma unroll
  for (int i = 0; i < 16; i += 2) {
    const int k = i;
    a[i]     = rp[k];
    a[i + 1] = rp[k + 1];
  }
  return a;
}

// B-matrix 32x16 (KxN) where B[k][n] = p[(col0+n)*stride + k]  (row-major in N)
DEVI v16h ldsBT_frag(const f16* p, int col0, int stride) {
  const int lane = threadIdx.x & 31;
  const int half = lane >> 4, lr = lane & 15;
  const f16* cp = p + (size_t)(col0 + lr) * stride + half * 16;
  v16h b;
#pragma unroll
  for (int i = 0; i < 16; ++i) b[i] = cp[i];
  return b;
}

// B-matrix 32x16 (KxN) where B[k][n] = p[k*stride + col0 + n]  (row-major in K)
DEVI v16h ldsB_frag(const f16* p, int col0, int stride) {
  const int lane = threadIdx.x & 31;
  const int half = lane >> 4, lr = lane & 15;
  const f16* cp = p + (size_t)(half * 16) * stride + col0 + lr;
  v16h b;
#pragma unroll
  for (int i = 0; i < 16; ++i) b[i] = cp[(size_t)i * stride];
  return b;
}

DEVI v8f wmma_f16(v16h a, v16h b, v8f c) {
  return __builtin_amdgcn_wmma_f32_16x16x32_f16(false, a, false, b, (short)0, c,
                                                false, false);
}

// ---------------- f32 -> f16 weight conversion ----------------
__global__ void cvt_kernel(const float* __restrict__ src, f16* __restrict__ dst, int n4) {
  const int i = blockIdx.x * blockDim.x + threadIdx.x;
  if (i < n4) {
    const float4 v = ((const float4*)src)[i];
    f16* d = dst + (size_t)i * 4;
    d[0] = (f16)v.x; d[1] = (f16)v.y; d[2] = (f16)v.z; d[3] = (f16)v.w;
  }
}

// ---------------- GEMM: Y[m][n] = sum_k A[m][k] * W[n][k] + bias[n] ----------------
// Block: 256 threads = 8 waves. Tile M=128 (16 rows/wave), N=64, K-step 32.
template <bool AF32, bool QKV_OUT>
__global__ __launch_bounds__(256)
void gemm_wmma(const void* __restrict__ Ap, const f16* __restrict__ Bw,
               const float* __restrict__ bias, void* __restrict__ Yp) {
  constexpr int K = HID, N = HID;
  constexpr int BM = 128, BK = 32;
  __shared__ __align__(16) f16 sA[BM * BK];   // 8 KB
  __shared__ __align__(16) f16 sB[64 * BK];   // 4 KB, stored [n][k]
  const int tid  = threadIdx.x;
  const int wave = tid >> 5;
  const int half = (tid >> 4) & 1, lr = tid & 15;
  const int m0 = blockIdx.y * BM;
  const int n0 = blockIdx.x * 64;

  v8f acc[4];
#pragma unroll
  for (int t = 0; t < 4; ++t) acc[t] = (v8f)0;

  for (int k0 = 0; k0 < K; k0 += BK) {
    // ---- W tile: async copy straight into LDS (ASYNCcnt) ----
    {
      const int n = tid >> 2, seg = tid & 3;    // 256 x b128 chunks
      async_ld_lds_b128(lds_off(sB + n * BK + seg * 8),
                        (unsigned)(((n0 + n) * K + k0 + seg * 8) * 2), Bw);
    }
    // ---- A tile ----
    if (AF32) {
      const float* A = (const float*)Ap;
#pragma unroll
      for (int j = 0; j < 4; ++j) {             // 1024 float4 chunks
        const int c = tid + 256 * j;
        const int row = c >> 3, seg = c & 7;
        const float* gp = A + (size_t)(m0 + row) * K + k0 + seg * 4;
        const float4 v = *(const float4*)gp;
        if (k0 + BK < K) __builtin_prefetch(gp + BK, 0, 0);   // global_prefetch_b8
        f16* d = sA + row * BK + seg * 4;
        d[0] = (f16)v.x; d[1] = (f16)v.y; d[2] = (f16)v.z; d[3] = (f16)v.w;
      }
    } else {
      const f16* A = (const f16*)Ap;
#pragma unroll
      for (int j = 0; j < 2; ++j) {             // 512 b128 chunks, async
        const int c = tid + 256 * j;
        const int row = c >> 2, seg = c & 3;
        async_ld_lds_b128(lds_off(sA + row * BK + seg * 8),
                          (unsigned)(((m0 + row) * K + k0 + seg * 8) * 2), A);
      }
    }
    wait_async0();
    __syncthreads();

    const v16h a = ldsA_frag(sA, wave * 16, BK);
#pragma unroll
    for (int t = 0; t < 4; ++t) {
      const v16h b = ldsBT_frag(sB, t * 16, BK);
      acc[t] = wmma_f16(a, b, acc[t]);
    }
    __syncthreads();
  }

#pragma unroll
  for (int t = 0; t < 4; ++t) {
    const int gn = n0 + t * 16 + lr;
    const float bv = bias[gn];
#pragma unroll
    for (int r = 0; r < 8; ++r) {
      const int gm = m0 + wave * 16 + r + 8 * half;
      const float val = acc[t][r] + bv;
      if (QKV_OUT) {   // write f16 into [B, NH, S, HD]
        const int b = gm >> 11, s = gm & (SEQ - 1);
        const int h = gn >> 6,  d = gn & (HDIM - 1);
        ((f16*)Yp)[((size_t)(b * NHEAD + h) * SEQ + s) * HDIM + d] = (f16)val;
      } else {          // f32 row-major [M, N]
        ((float*)Yp)[(size_t)gm * N + gn] = val;
      }
    }
  }
}

// ---------------- Flash attention (causal) ----------------
// Grid: (B*NH, S/128). 8 waves, each owns a 16-row q tile. Key blocks of 32.
__global__ __launch_bounds__(256)
void attn_wmma(const f16* __restrict__ Q, const f16* __restrict__ Kk,
               const f16* __restrict__ V, f16* __restrict__ ctx) {
  __shared__ __align__(16) f16 smem[128 * 64];   // 16 KB
  f16* sK = smem;                 // 32 x 64
  f16* sV = smem + 32 * 64;       // 32 x 64
  f16* sP = smem + 64 * 64;       // 8 waves x (16 x 32)

  const int tid  = threadIdx.x;
  const int wave = tid >> 5;
  const int half = (tid >> 4) & 1, lr = tid & 15;
  const int bh    = blockIdx.x;
  const int qbase = blockIdx.y * 128;
  const int qrow0 = qbase + wave * 16;

  const size_t hoff = (size_t)bh * SEQ * HDIM;
  const f16* Qh = Q  + hoff;
  const f16* Kh = Kk + hoff;
  const f16* Vh = V  + hoff;

  // Stage Q block (128x64) via async copies, then pull per-wave A fragments
#pragma unroll
  for (int j = 0; j < 4; ++j) {                    // 1024 b128 chunks
    const int c = tid + 256 * j;
    const int row = c >> 3, seg = c & 7;
    async_ld_lds_b128(lds_off(smem + row * 64 + seg * 8),
                      (unsigned)((((qbase + row) * HDIM) + seg * 8) * 2), Qh);
  }
  wait_async0();
  __syncthreads();
  const v16h aq0 = ldsA_frag(smem,      wave * 16, 64);   // K = 0..31
  const v16h aq1 = ldsA_frag(smem + 32, wave * 16, 64);   // K = 32..63
  __syncthreads();

  float mrow[8], lrow[8];
  v8f O[4];
#pragma unroll
  for (int r = 0; r < 8; ++r) { mrow[r] = -1e9f; lrow[r] = 0.f; }
#pragma unroll
  for (int t = 0; t < 4; ++t) O[t] = (v8f)0;

  const int nkb = qbase / 32 + 4;   // causal extent of this 128-row block
  for (int kb_i = 0; kb_i < nkb; ++kb_i) {
    const int kb = kb_i * 32;
    __syncthreads();
    {                                              // K + V tiles: async b128 each
      const int row = tid >> 3, seg = tid & 7;     // 32 rows x 8 segs
      const unsigned goff = (unsigned)((((kb + row) * HDIM) + seg * 8) * 2);
      async_ld_lds_b128(lds_off(sK + row * 64 + seg * 8), goff, Kh);
      async_ld_lds_b128(lds_off(sV + row * 64 + seg * 8), goff, Vh);
    }
    wait_async0();
    __syncthreads();

    // scores: S = Q . K^T   (two 16x16 tiles, K-dim 64 = 2 WMMA each)
    v8f sc[2];
#pragma unroll
    for (int t = 0; t < 2; ++t) {
      sc[t] = (v8f)0;
      const v16h b0 = ldsBT_frag(sK,      t * 16, 64);
      const v16h b1 = ldsBT_frag(sK + 32, t * 16, 64);
      sc[t] = wmma_f16(aq0, b0, sc[t]);
      sc[t] = wmma_f16(aq1, b1, sc[t]);
    }
    // scale 1/sqrt(64) + causal mask (matches reference -1e9)
#pragma unroll
    for (int t = 0; t < 2; ++t)
#pragma unroll
      for (int r = 0; r < 8; ++r) {
        const int q  = qrow0 + r + 8 * half;
        const int kx = kb + t * 16 + lr;
        const float v = sc[t][r] * 0.125f;
        sc[t][r] = (kx > q) ? -1e9f : v;
      }
    // online softmax: row M = r + 8*half lives in one 16-lane group
#pragma unroll
    for (int r = 0; r < 8; ++r) {
      float x = fmaxf(sc[0][r], sc[1][r]);
#pragma unroll
      for (int off = 1; off < 16; off <<= 1) x = fmaxf(x, __shfl_xor(x, off, 32));
      const float mnew = fmaxf(mrow[r], x);
      const float corr = __expf(mrow[r] - mnew);
      const float p0 = __expf(sc[0][r] - mnew);
      const float p1 = __expf(sc[1][r] - mnew);
      float rs = p0 + p1;
#pragma unroll
      for (int off = 1; off < 16; off <<= 1) rs += __shfl_xor(rs, off, 32);
      lrow[r] = lrow[r] * corr + rs;
      mrow[r] = mnew;
      sc[0][r] = p0; sc[1][r] = p1;
#pragma unroll
      for (int t = 0; t < 4; ++t) O[t][r] *= corr;
    }
    // re-layout P (C-frag -> A-frag) via wave-private LDS; in-order per wave
    f16* sPw = sP + wave * (16 * 32);
#pragma unroll
    for (int t = 0; t < 2; ++t)
#pragma unroll
      for (int r = 0; r < 8; ++r)
        sPw[(r + 8 * half) * 32 + t * 16 + lr] = (f16)sc[t][r];
    const v16h ap = ldsA_frag(sPw, 0, 32);
#pragma unroll
    for (int t = 0; t < 4; ++t) {
      const v16h bv = ldsB_frag(sV, t * 16, 64);
      O[t] = wmma_f16(ap, bv, O[t]);
    }
  }

  // epilogue: ctx[b, s, h*64 + d] = O / l  (f16)
  const int b = bh >> 4, h = bh & 15;
#pragma unroll
  for (int t = 0; t < 4; ++t)
#pragma unroll
    for (int r = 0; r < 8; ++r) {
      const int s = qrow0 + r + 8 * half;
      const float val = O[t][r] / lrow[r];
      ctx[(size_t)(b * SEQ + s) * HID + h * HDIM + t * 16 + lr] = (f16)val;
    }
}

// ---------------- host launch ----------------
extern "C" void kernel_launch(void* const* d_in, const int* in_sizes, int n_in,
                              void* d_out, int out_size, void* d_ws, size_t ws_size,
                              hipStream_t stream) {
  (void)in_sizes; (void)n_in; (void)out_size; (void)ws_size;
  const float* x  = (const float*)d_in[0];
  // d_in[1] = mask (unused: causal mask applied analytically)
  const float* Wq = (const float*)d_in[2];
  const float* bq = (const float*)d_in[3];
  const float* Wk = (const float*)d_in[4];
  const float* bk = (const float*)d_in[5];
  const float* Wv = (const float*)d_in[6];
  const float* bv = (const float*)d_in[7];
  const float* Wo = (const float*)d_in[8];
  const float* bo = (const float*)d_in[9];

  // workspace layout (f16 elems): 4 weights (4M) + Q,K,V,ctx (4 x 8M) = 72 MB
  f16* ws = (f16*)d_ws;
  const size_t WSZ = (size_t)HID * HID;        // 1M elems per weight
  const size_t TSZ = (size_t)MTOT * HID;       // 8M elems per tensor
  f16* Wq16 = ws;
  f16* Wk16 = ws + WSZ;
  f16* Wv16 = ws + 2 * WSZ;
  f16* Wo16 = ws + 3 * WSZ;
  f16* Qb   = ws + 4 * WSZ;
  f16* Kb   = Qb + TSZ;
  f16* Vb   = Kb + TSZ;
  f16* Cx   = Vb + TSZ;

  const int n4 = (HID * HID) / 4;
  const dim3 cg((n4 + 255) / 256), cb(256);
  cvt_kernel<<<cg, cb, 0, stream>>>(Wq, Wq16, n4);
  cvt_kernel<<<cg, cb, 0, stream>>>(Wk, Wk16, n4);
  cvt_kernel<<<cg, cb, 0, stream>>>(Wv, Wv16, n4);
  cvt_kernel<<<cg, cb, 0, stream>>>(Wo, Wo16, n4);

  const dim3 gg(HID / 64, MTOT / 128);         // (16, 64)
  gemm_wmma<true,  true ><<<gg, 256, 0, stream>>>(x,  Wq16, bq, Qb);
  gemm_wmma<true,  true ><<<gg, 256, 0, stream>>>(x,  Wk16, bk, Kb);
  gemm_wmma<true,  true ><<<gg, 256, 0, stream>>>(x,  Wv16, bv, Vb);

  attn_wmma<<<dim3(BATCH * NHEAD, SEQ / 128), 256, 0, stream>>>(Qb, Kb, Vb, Cx);

  gemm_wmma<false, false><<<gg, 256, 0, stream>>>(Cx, Wo16, bo, d_out);
}
